// CollaborativeWaterfallMoE_74105365725738
// MI455X (gfx1250) — matrix operand order repaired and checked by reading, not compiled, not run on hardware
//
#include <hip/hip_runtime.h>
#include <hip/hip_bf16.h>
#include <math.h>

// ============================================================================
// CollaborativeWaterfallMoE for MI455X (gfx1250, wave32, WMMA 16x16x32 f16)
// Fused per-image expert CNN with halo-padded LDS activations (branchless
// B-fragments via immediate-offset DS loads), conflict-avoiding channel
// strides, f32 accumulation, vectorized LDS/global data movement.
// ============================================================================

typedef __attribute__((ext_vector_type(16))) _Float16 v16h;
typedef __attribute__((ext_vector_type(16), aligned(16))) _Float16 v16h_a16;
typedef __attribute__((ext_vector_type(8)))  _Float16 v8h;
typedef __attribute__((ext_vector_type(8)))  float    v8f;
using half_t = _Float16;

#define NE 4
#define BTOT 4096
#define CCAP 1024
#define TINV 10.0f
#define BNEPS 1e-5f

// channel strides (halves): padded for LDS bank-conflict avoidance + 16B align
#define CS1 72      // 64ch + 8 pad  (144B = 36-bank stride)
#define CS4 136     // 128ch + 8 pad (272B = 4-bank stride)

__device__ inline v8f wmma16(v16h a, v16h b, v8f c) {
  return __builtin_amdgcn_wmma_f32_16x16x32_f16(false, a, false, b, (short)0, c,
                                                false, false);
}

// 32B B-fragment as one vector load (16B-aligned; lowers to 2x ds_load_b128
// into consecutive VGPRs, no repacking moves)
__device__ inline v16h ld_bfrag(const half_t* p) {
  return *(const v16h_a16*)p;
}

__device__ inline void border_coords(int bp, int P, int& y, int& x) {
  if (bp < P)              { y = 0;                x = bp; }
  else if (bp < 2 * P)     { y = P - 1;            x = bp - P; }
  else if (bp < 3 * P - 2) { y = bp - 2 * P + 1;   x = 0; }
  else                     { y = bp - (3 * P - 2) + 1; x = P - 1; }
}

// ---------------------------------------------------------------------------
// Prep: fold BN+bias into per-oc (scale, bias).
// ---------------------------------------------------------------------------
__global__ void prep_bn_kernel(const float* __restrict__ g, const float* __restrict__ be,
                               const float* __restrict__ m, const float* __restrict__ v,
                               const float* __restrict__ b,
                               float* __restrict__ scale, float* __restrict__ bias, int C) {
  int i = blockIdx.x * blockDim.x + threadIdx.x;
  if (i >= NE * C) return;
  float s = g[i] * rsqrtf(v[i] + BNEPS);
  int e = i / C, c = i % C;
  scale[e * 256 + c] = s;
  bias[e * 256 + c] = (be[i] - m[i] * s) + b[i] * s;
}

// ---------------------------------------------------------------------------
// Pack conv weights into WMMA A-fragment order (f16).
// A 16x32 f16, this code's contiguous k order: lane L: m=L%16; half h -> k = h+16*(L>=16)
// ---------------------------------------------------------------------------
__global__ void pack_conv_kernel(const float* __restrict__ w, half_t* __restrict__ dst,
                                 int OC, int IC) {
  int frag = blockIdx.x;
  int nOct = OC / 16, nCh = IC / 32;
  int e   = frag / (nOct * 9 * nCh);
  int r   = frag % (nOct * 9 * nCh);
  int oct = r / (9 * nCh);
  int r2  = r % (9 * nCh);
  int pos = r2 / nCh;
  int ch  = r2 % nCh;
  for (int t = threadIdx.x; t < 512; t += blockDim.x) {
    int lane = t >> 4, h = t & 15;
    int mrow = lane & 15;
    int k = h + 16 * (lane >> 4);
    int oc = oct * 16 + mrow;
    int ic = ch * 32 + k;
    dst[(size_t)frag * 512 + t] = (half_t)w[(((size_t)e * OC + oc) * IC + ic) * 9 + pos];
  }
}

__global__ void pack_conv1_kernel(const float* __restrict__ w1, half_t* __restrict__ dst) {
  int frag = blockIdx.x;                 // E*4
  int e = frag >> 2, oct = frag & 3;
  for (int t = threadIdx.x; t < 512; t += blockDim.x) {
    int lane = t >> 4, h = t & 15;
    int mrow = lane & 15;
    int k = h + 16 * (lane >> 4);
    float val = 0.f;
    if (k < 27) {
      int ic = k / 9, pos = k % 9;
      int oc = oct * 16 + mrow;
      val = w1[((size_t)(e * 64 + oc) * 3 + ic) * 9 + pos];
    }
    dst[(size_t)frag * 512 + t] = (half_t)val;
  }
}

// ---------------------------------------------------------------------------
// Scores + gumbel
// ---------------------------------------------------------------------------
__global__ void scores_kernel(const float* __restrict__ x, const float* __restrict__ ws1,
                              const float* __restrict__ bs1, const float* __restrict__ ws2,
                              const float* __restrict__ bs2, const float* __restrict__ gu,
                              float* __restrict__ noisy) {
  __shared__ float red[16][128];
  int img = blockIdx.x;
  const float* xi = x + (size_t)img * 3072;
  float acc[16];
#pragma unroll
  for (int i = 0; i < 16; ++i) acc[i] = 0.f;
  for (int p = threadIdx.x; p < 1024; p += 128) {
    float c0 = xi[p], c1 = xi[1024 + p], c2 = xi[2048 + p];
#pragma unroll
    for (int eo = 0; eo < 16; ++eo) {
      const float* w = ws1 + eo * 3;
      float v = c0 * w[0] + c1 * w[1] + c2 * w[2] + bs1[eo];
      acc[eo] += fmaxf(v, 0.f);
    }
  }
#pragma unroll
  for (int i = 0; i < 16; ++i) red[i][threadIdx.x] = acc[i];
  __syncthreads();
  for (int s = 64; s > 0; s >>= 1) {
    if (threadIdx.x < (unsigned)s)
#pragma unroll
      for (int i = 0; i < 16; ++i) red[i][threadIdx.x] += red[i][threadIdx.x + s];
    __syncthreads();
  }
  if (threadIdx.x < NE) {
    int e = threadIdx.x;
    float sc = bs2[e];
#pragma unroll
    for (int o = 0; o < 4; ++o) sc += (red[e * 4 + o][0] * (1.f / 1024.f)) * ws2[e * 4 + o];
    float u = gu[img * NE + e];
    float gum = -logf(-logf(u));
    noisy[img * NE + e] = (sc + gum) * TINV;
  }
}

// ---------------------------------------------------------------------------
// Waterfall routing (numpy-exact order). Single block.
// ---------------------------------------------------------------------------
__global__ void route_kernel(const float* __restrict__ noisy, int* __restrict__ eid,
                             int* __restrict__ rem, int* __restrict__ rem2,
                             int* __restrict__ best) {
  __shared__ int s_nrem;
  __shared__ int s_cap[NE];
  if (threadIdx.x == 0) {
    s_nrem = BTOT;
    for (int e = 0; e < NE; ++e) s_cap[e] = 0;
  }
  for (int i = threadIdx.x; i < BTOT; i += blockDim.x) rem[i] = i;
  __syncthreads();
  int quota = 1;
  for (int it = 0; it < 32; ++it) {
    int nrem = s_nrem;
    if (nrem == 0) break;
    float fac[NE];
    bool full[NE];
#pragma unroll
    for (int e = 0; e < NE; ++e) {
      int c = s_cap[e];
      fac[e] = 1.f - fminf((float)c * (1.0f / (float)CCAP), 1.f);
      full[e] = (c >= CCAP);
    }
    for (int i = threadIdx.x; i < nrem; i += blockDim.x) {
      int b = rem[i];
      float bv = -INFINITY;
      int bi = 0;
#pragma unroll
      for (int e = 0; e < NE; ++e) {
        float s = full[e] ? -INFINITY : noisy[b * NE + e] * fac[e];
        if (s > bv) { bv = s; bi = e; }
      }
      best[i] = bi;
    }
    __syncthreads();
    if (threadIdx.x == 0) {
      int space[NE], taken[NE];
      for (int e = 0; e < NE; ++e) {
        int sp = CCAP - s_cap[e];
        if (sp > quota) sp = quota;
        if (sp < 0) sp = 0;
        space[e] = sp;
        taken[e] = 0;
      }
      int nn = 0;
      for (int i = 0; i < nrem; ++i) {
        int e = best[i];
        if (taken[e] < space[e]) { eid[rem[i]] = e; taken[e]++; }
        else rem2[nn++] = rem[i];
      }
      for (int e = 0; e < NE; ++e) s_cap[e] += taken[e];
      s_nrem = nn;
    }
    __syncthreads();
    int nn = s_nrem;
    for (int i = threadIdx.x; i < nn; i += blockDim.x) rem[i] = rem2[i];
    __syncthreads();
    quota = (quota >= CCAP) ? CCAP : (quota << 1);
  }
}

// ---------------------------------------------------------------------------
// Fused conv1 -> LDS(34x34 halo, CS1 stride), conv2 + pool -> global [p][c].
// One workgroup per image, 256 threads = 8 waves. ~174KB LDS (of 320KB/WGP).
// ---------------------------------------------------------------------------
__global__ __launch_bounds__(256)
void fused12_kernel(const float* __restrict__ x, const half_t* __restrict__ W1P,
                    const half_t* __restrict__ W2P,
                    const float* __restrict__ sc1, const float* __restrict__ bi1,
                    const float* __restrict__ sc2, const float* __restrict__ bi2,
                    const int* __restrict__ eid, half_t* __restrict__ actA) {
  __shared__ __attribute__((aligned(32))) half_t xpad[3 * 34 * 34];     // [ic][34][34]
  __shared__ __attribute__((aligned(32))) half_t act1p[34 * 34 * CS1];  // [py][px][CS1]
  __shared__ float s1l[64], b1l[64], s2l[64], b2l[64];
  int img = blockIdx.x;
  int e = eid[img];
  int tid = threadIdx.x;
  int lane = tid & 31, wv = tid >> 5;
  int hi = lane >> 4, lm = lane & 15;

  v8h z8;
#pragma unroll
  for (int i = 0; i < 8; ++i) z8[i] = (half_t)0.0f;

  if (tid < 64) {
    s1l[tid] = sc1[e * 256 + tid];
    b1l[tid] = bi1[e * 256 + tid];
    s2l[tid] = sc2[e * 256 + tid];
    b2l[tid] = bi2[e * 256 + tid];
  }
  // zero halos
  for (int i = tid; i < 3 * 132; i += 256) {
    int ic = i / 132, bp = i % 132, y, xx;
    border_coords(bp, 34, y, xx);
    xpad[ic * 1156 + y * 34 + xx] = (half_t)0.0f;
  }
  for (int i = tid; i < 132 * 9; i += 256) {
    int bp = i / 9, ck = i % 9, y, xx;
    border_coords(bp, 34, y, xx);
    *(v8h*)(act1p + (y * 34 + xx) * CS1 + ck * 8) = z8;
  }
  // x -> f16 interior
  const float* xi = x + (size_t)img * 3072;
  for (int i = tid; i < 3072; i += 256) {
    int ic = i >> 10, p = i & 1023;
    int y = p >> 5, xx = p & 31;
    xpad[ic * 1156 + (y + 1) * 34 + (xx + 1)] = (half_t)xi[i];
  }
  __syncthreads();

  // hoisted conv1 gather offsets (per lane, branchless inner loop)
  int offk[16];
#pragma unroll
  for (int h = 0; h < 16; ++h) {
    int k = h + 16 * hi;
    int ic = k / 9, r9 = k % 9;
    int off = ic * 1156 + (r9 / 3) * 34 + (r9 % 3);
    offk[h] = (k < 27) ? off : 0;     // k>=27 reads ic0 tap (0,0); its packed weight is 0
  }

  // ---- conv1: 4 octiles x 64 pixel-tiles ----
  for (int task = wv; task < 256; task += 8) {
    int oct = task >> 6;
    int pxt = task & 63;
    int pix = pxt * 16 + lm;
    int py = pix >> 5, px = pix & 31;
    int base = py * 34 + px;
    v16h bfrag;
#pragma unroll
    for (int h = 0; h < 16; ++h) bfrag[h] = xpad[offk[h] + base];
    v16h afrag = *(const v16h*)(W1P + (size_t)(e * 4 + oct) * 512 + lane * 16);
    v8f acc;
#pragma unroll
    for (int q = 0; q < 8; ++q) acc[q] = 0.f;
    acc = wmma16(afrag, bfrag, acc);
    int ocb = oct * 16 + 8 * hi;
    v8h o8;
#pragma unroll
    for (int q = 0; q < 8; ++q)
      o8[q] = (half_t)fmaxf(acc[q] * s1l[ocb + q] + b1l[ocb + q], 0.f);
    *(v8h*)(act1p + ((py + 1) * 34 + (px + 1)) * CS1 + ocb) = o8;
  }
  __syncthreads();

  // ---- conv2 + maxpool: 4 octiles x 16 pooled-rows x 2 xsegs ----
  const half_t* w2base = W2P + (size_t)(e * 4) * 18 * 512;
  half_t* actAi = actA + (size_t)img * 64 * 256;     // [p(16x16)][c64]
  for (int task = wv; task < 128; task += 8) {
    int oct = task >> 5;
    int r = task & 31;
    int prow = r >> 1, xseg = r & 1;
    int y0 = prow * 2;
    int xc = xseg * 16 + lm;
    const half_t* bbase = act1p + (y0 * 34 + xc) * CS1 + 16 * hi;  // halo top-left
    const half_t* wf = w2base + (size_t)oct * 18 * 512 + lane * 16;
    __builtin_prefetch(wf, 0, 3);
    v8f acc0, acc1;
#pragma unroll
    for (int q = 0; q < 8; ++q) { acc0[q] = 0.f; acc1[q] = 0.f; }
#pragma unroll
    for (int pos = 0; pos < 9; ++pos) {
      const int dy = pos / 3, dx = pos % 3;       // halo-relative (0..2)
#pragma unroll
      for (int ch = 0; ch < 2; ++ch) {
        v16h afrag = *(const v16h*)(wf + (size_t)(pos * 2 + ch) * 512);
        v16h bf0 = ld_bfrag(bbase + (dy * 34 + dx) * CS1 + ch * 32);
        v16h bf1 = ld_bfrag(bbase + ((dy + 1) * 34 + dx) * CS1 + ch * 32);
        acc0 = wmma16(afrag, bf0, acc0);
        acc1 = wmma16(afrag, bf1, acc1);
      }
    }
    int ocb = oct * 16 + 8 * hi;
    v8h p8;
#pragma unroll
    for (int q = 0; q < 8; ++q) {
      float v0 = fmaxf(acc0[q] * s2l[ocb + q] + b2l[ocb + q], 0.f);
      float v1 = fmaxf(acc1[q] * s2l[ocb + q] + b2l[ocb + q], 0.f);
      float mm = fmaxf(v0, v1);
      mm = fmaxf(mm, __shfl_xor(mm, 1, 32));
      p8[q] = (half_t)mm;
    }
    if ((lm & 1) == 0)    // even lanes hold pooled pixel (xc>>1): one 16B store
      *(v8h*)(actAi + (prow * 16 + (xc >> 1)) * 64 + ocb) = p8;
  }
}

// ---------------------------------------------------------------------------
// Fused conv3 + pool (16x16 -> 8x8) and conv4 + global avg pool. ~78KB LDS.
// ---------------------------------------------------------------------------
__global__ __launch_bounds__(256)
void fused34_kernel(const half_t* __restrict__ actA, const half_t* __restrict__ W3P,
                    const half_t* __restrict__ W4P,
                    const float* __restrict__ sc3, const float* __restrict__ bi3,
                    const float* __restrict__ sc4, const float* __restrict__ bi4,
                    const int* __restrict__ eid, float* __restrict__ feat) {
  __shared__ __attribute__((aligned(32))) half_t a3p[18 * 18 * CS1];  // 16x16 + halo
  __shared__ __attribute__((aligned(32))) half_t a4p[10 * 10 * CS4];  // 8x8 + halo
  __shared__ float s3l[128], b3l[128], s4l[256], b4l[256];
  int img = blockIdx.x;
  int e = eid[img];
  int tid = threadIdx.x;
  int lane = tid & 31, wv = tid >> 5;
  int hi = lane >> 4, lm = lane & 15;

  v8h z8;
#pragma unroll
  for (int i = 0; i < 8; ++i) z8[i] = (half_t)0.0f;

  if (tid < 128) { s3l[tid] = sc3[e * 256 + tid]; b3l[tid] = bi3[e * 256 + tid]; }
  s4l[tid] = sc4[e * 256 + tid];
  b4l[tid] = bi4[e * 256 + tid];
  for (int i = tid; i < 68 * 9; i += 256) {
    int bp = i / 9, ck = i % 9, y, xx;
    border_coords(bp, 18, y, xx);
    *(v8h*)(a3p + (y * 18 + xx) * CS1 + ck * 8) = z8;
  }
  for (int i = tid; i < 36 * 17; i += 256) {
    int bp = i / 17, ck = i % 17, y, xx;
    border_coords(bp, 10, y, xx);
    *(v8h*)(a4p + (y * 10 + xx) * CS4 + ck * 8) = z8;
  }
  // global [p][c] -> LDS interior (vectorized 16B copies, coalesced reads)
  const half_t* Ai = actA + (size_t)img * 64 * 256;
  for (int i = tid; i < 2048; i += 256) {
    int p = i >> 3, ck = i & 7;
    int y = p >> 4, xx = p & 15;
    *(v8h*)(a3p + ((y + 1) * 18 + (xx + 1)) * CS1 + ck * 8) =
        *(const v8h*)(Ai + p * 64 + ck * 8);
  }
  __syncthreads();

  // ---- conv3 + pool: 8 octiles x 8 pooled-rows ----
  const half_t* w3base = W3P + (size_t)(e * 8) * 18 * 512;
  for (int task = wv; task < 64; task += 8) {
    int oct = task >> 3, prow = task & 7;
    int y0 = prow * 2;
    const half_t* bbase = a3p + (y0 * 18 + lm) * CS1 + 16 * hi;
    const half_t* wf = w3base + (size_t)oct * 18 * 512 + lane * 16;
    __builtin_prefetch(wf, 0, 3);
    v8f acc0, acc1;
#pragma unroll
    for (int q = 0; q < 8; ++q) { acc0[q] = 0.f; acc1[q] = 0.f; }
#pragma unroll
    for (int pos = 0; pos < 9; ++pos) {
      const int dy = pos / 3, dx = pos % 3;
#pragma unroll
      for (int ch = 0; ch < 2; ++ch) {
        v16h afrag = *(const v16h*)(wf + (size_t)(pos * 2 + ch) * 512);
        v16h bf0 = ld_bfrag(bbase + (dy * 18 + dx) * CS1 + ch * 32);
        v16h bf1 = ld_bfrag(bbase + ((dy + 1) * 18 + dx) * CS1 + ch * 32);
        acc0 = wmma16(afrag, bf0, acc0);
        acc1 = wmma16(afrag, bf1, acc1);
      }
    }
    int ocb = oct * 16 + 8 * hi;
    v8h p8;
#pragma unroll
    for (int q = 0; q < 8; ++q) {
      float v0 = fmaxf(acc0[q] * s3l[ocb + q] + b3l[ocb + q], 0.f);
      float v1 = fmaxf(acc1[q] * s3l[ocb + q] + b3l[ocb + q], 0.f);
      float mm = fmaxf(v0, v1);
      mm = fmaxf(mm, __shfl_xor(mm, 1, 32));
      p8[q] = (half_t)mm;
    }
    if ((lm & 1) == 0)
      *(v8h*)(a4p + ((prow + 1) * 10 + (lm >> 1) + 1) * CS4 + ocb) = p8;
  }
  __syncthreads();

  // ---- conv4 + GAP: 16 octiles, two independent WMMA chains ----
  const half_t* w4base = W4P + (size_t)(e * 16) * 36 * 512;
  for (int oct = wv; oct < 16; oct += 8) {
    const half_t* wf = w4base + (size_t)oct * 36 * 512 + lane * 16;
    __builtin_prefetch(wf, 0, 3);
    float sum8[8];
#pragma unroll
    for (int q = 0; q < 8; ++q) sum8[q] = 0.f;
    int ocb = oct * 16 + 8 * hi;
    for (int nt = 0; nt < 4; ++nt) {
      int pix = nt * 16 + lm;
      int py = pix >> 3, px = pix & 7;
      const half_t* bbase = a4p + (py * 10 + px) * CS4 + 16 * hi;
      v8f accA, accB;
#pragma unroll
      for (int q = 0; q < 8; ++q) { accA[q] = 0.f; accB[q] = 0.f; }
#pragma unroll
      for (int pos = 0; pos < 9; ++pos) {
        const int dy = pos / 3, dx = pos % 3;
        const half_t* bp = bbase + (dy * 10 + dx) * CS4;
        accA = wmma16(*(const v16h*)(wf + (size_t)(pos * 4 + 0) * 512),
                      ld_bfrag(bp + 0 * 32), accA);
        accB = wmma16(*(const v16h*)(wf + (size_t)(pos * 4 + 1) * 512),
                      ld_bfrag(bp + 1 * 32), accB);
        accA = wmma16(*(const v16h*)(wf + (size_t)(pos * 4 + 2) * 512),
                      ld_bfrag(bp + 2 * 32), accA);
        accB = wmma16(*(const v16h*)(wf + (size_t)(pos * 4 + 3) * 512),
                      ld_bfrag(bp + 3 * 32), accB);
      }
#pragma unroll
      for (int q = 0; q < 8; ++q)
        sum8[q] += fmaxf((accA[q] + accB[q]) * s4l[ocb + q] + b4l[ocb + q], 0.f);
    }
#pragma unroll
    for (int q = 0; q < 8; ++q) {
      float v = sum8[q];
      v += __shfl_xor(v, 1, 32);
      v += __shfl_xor(v, 2, 32);
      v += __shfl_xor(v, 4, 32);
      v += __shfl_xor(v, 8, 32);
      sum8[q] = v;
    }
    if (lm == 0) {
#pragma unroll
      for (int q = 0; q < 8; ++q)
        feat[(size_t)img * 256 + ocb + q] = sum8[q] * (1.f / 64.f);
    }
  }
}

// ---------------------------------------------------------------------------
// Head (scalar, 560 MFLOP total)
// ---------------------------------------------------------------------------
__global__ __launch_bounds__(256)
void head_kernel(const float* __restrict__ feat, const float* __restrict__ pw,
                 const float* __restrict__ pb, const float* __restrict__ cw,
                 const float* __restrict__ cb, const int* __restrict__ eid,
                 float* __restrict__ out) {
  __shared__ float z[256];
  __shared__ float z2[256];
  int img = blockIdx.x;
  int e = eid[img];
  int t = threadIdx.x;
  z[t] = feat[(size_t)img * 256 + t];
  __syncthreads();
  const float* pwr = pw + ((size_t)e * 256 + t) * 256;
  float acc = pb[e * 256 + t];
  for (int k = 0; k < 256; ++k) acc += pwr[k] * z[k];
  z2[t] = acc;
  __syncthreads();
  if (t < 10) {
    const float* cwr = cw + ((size_t)e * 10 + t) * 256;
    float a = cb[e * 10 + t];
    for (int k = 0; k < 256; ++k) a += cwr[k] * z2[k];
    out[(size_t)img * 10 + t] = a;
  }
}

// ---------------------------------------------------------------------------
extern "C" void kernel_launch(void* const* d_in, const int* in_sizes, int n_in,
                              void* d_out, int out_size, void* d_ws, size_t ws_size,
                              hipStream_t stream) {
  (void)in_sizes; (void)n_in; (void)out_size; (void)ws_size;
  const float* x   = (const float*)d_in[0];
  const float* w1  = (const float*)d_in[1];
  const float* w2  = (const float*)d_in[7];
  const float* w3  = (const float*)d_in[13];
  const float* w4  = (const float*)d_in[19];
  const float* pw  = (const float*)d_in[25];
  const float* pb  = (const float*)d_in[26];
  const float* cw  = (const float*)d_in[27];
  const float* cb  = (const float*)d_in[28];
  const float* ws1 = (const float*)d_in[29];
  const float* bs1 = (const float*)d_in[30];
  const float* ws2 = (const float*)d_in[31];
  const float* bs2 = (const float*)d_in[32];
  const float* gu  = (const float*)d_in[33];
  float* out = (float*)d_out;

  char* base = (char*)d_ws;
  size_t off = 0;
  auto take = [&](size_t bytes) -> void* {
    off = (off + 255) & ~(size_t)255;
    void* p = base + off;
    off += bytes;
    return p;
  };
  float* noisy = (float*)take((size_t)BTOT * NE * 4);
  int* eid  = (int*)take((size_t)BTOT * 4);
  int* rem  = (int*)take((size_t)BTOT * 4);
  int* rem2 = (int*)take((size_t)BTOT * 4);
  int* best = (int*)take((size_t)BTOT * 4);
  float* scAll = (float*)take((size_t)4 * NE * 256 * 4);
  float* biAll = (float*)take((size_t)4 * NE * 256 * 4);
  half_t* W1P = (half_t*)take((size_t)NE * 4 * 512 * 2);
  half_t* W2P = (half_t*)take((size_t)NE * 4 * 18 * 512 * 2);
  half_t* W3P = (half_t*)take((size_t)NE * 8 * 18 * 512 * 2);
  half_t* W4P = (half_t*)take((size_t)NE * 16 * 36 * 512 * 2);
  half_t* actA = (half_t*)take((size_t)BTOT * 64 * 256 * 2);   // 128 MB
  float* feat = (float*)take((size_t)BTOT * 256 * 4);          // 4 MB

  const int Cl[4] = {64, 64, 128, 256};
  for (int l = 0; l < 4; ++l) {
    const float* g  = (const float*)d_in[1 + 6 * l + 2];
    const float* be = (const float*)d_in[1 + 6 * l + 3];
    const float* m  = (const float*)d_in[1 + 6 * l + 4];
    const float* v  = (const float*)d_in[1 + 6 * l + 5];
    const float* b  = (const float*)d_in[1 + 6 * l + 1];
    int n = NE * Cl[l];
    prep_bn_kernel<<<(n + 255) / 256, 256, 0, stream>>>(
        g, be, m, v, b, scAll + l * NE * 256, biAll + l * NE * 256, Cl[l]);
  }

  pack_conv1_kernel<<<NE * 4, 256, 0, stream>>>(w1, W1P);
  pack_conv_kernel<<<NE * 4 * 9 * 2, 256, 0, stream>>>(w2, W2P, 64, 64);
  pack_conv_kernel<<<NE * 8 * 9 * 2, 256, 0, stream>>>(w3, W3P, 128, 64);
  pack_conv_kernel<<<NE * 16 * 9 * 4, 256, 0, stream>>>(w4, W4P, 256, 128);

  scores_kernel<<<BTOT, 128, 0, stream>>>(x, ws1, bs1, ws2, bs2, gu, noisy);
  route_kernel<<<1, 256, 0, stream>>>(noisy, eid, rem, rem2, best);

  fused12_kernel<<<BTOT, 256, 0, stream>>>(
      x, W1P, W2P, scAll + 0 * NE * 256, biAll + 0 * NE * 256,
      scAll + 1 * NE * 256, biAll + 1 * NE * 256, eid, actA);
  fused34_kernel<<<BTOT, 256, 0, stream>>>(
      actA, W3P, W4P, scAll + 2 * NE * 256, biAll + 2 * NE * 256,
      scAll + 3 * NE * 256, biAll + 3 * NE * 256, eid, feat);
  head_kernel<<<BTOT, 256, 0, stream>>>(feat, pw, pb, cw, cb, eid, out);
}
